// SelfAttention_V2_26431228739710
// MI455X (gfx1250) — matrix operand order repaired and checked by reading, not compiled, not run on hardware
//
#include <hip/hip_runtime.h>

#define N_TOK 4096
#define D_DIM 1024

typedef __attribute__((ext_vector_type(16))) __bf16        v16bf;
typedef __attribute__((ext_vector_type(8)))  float         v8f;
typedef __attribute__((ext_vector_type(4)))  unsigned int  v4u;
typedef __attribute__((ext_vector_type(8)))  unsigned int  v8u;

__device__ __forceinline__ unsigned short f2bf(float f) {
    unsigned int u = __float_as_uint(f);
    u += 0x7FFFu + ((u >> 16) & 1u);   // round-to-nearest-even
    return (unsigned short)(u >> 16);
}
__device__ __forceinline__ unsigned int pack2(float a, float b) {
    return (unsigned int)f2bf(a) | ((unsigned int)f2bf(b) << 16);
}

// ---- Fragment-native ("swizzled") storage -----------------------------------
// One 16x16x32 bf16 operand fragment = 512 elements stored as:
//   idx = frag_base + (run*32 + lane)*8 + e8      (run = element_index>>3)
// Each lane owns two contiguous 16B runs (2x b128 loads); runs are contiguous
// across lanes (512B) so global accesses are perfectly coalesced.
//
// A-operand element (m,k):  lane=(m&15)+16*((k>>3)&1), run=(k>>4)&1, e8=k&7
// B-operand element (k,n):  lane=(n&15)+16*((k>>3)&1), run=(k>>4)&1, e8=k&7
__device__ __forceinline__ v16bf load_frag_sw(const unsigned short* base, int lane) {
    v4u a = *(const v4u*)(base + lane * 8);
    v4u b = *(const v4u*)(base + 256 + lane * 8);
    v8u c;
    c[0] = a[0]; c[1] = a[1]; c[2] = a[2]; c[3] = a[3];
    c[4] = b[0]; c[5] = b[1]; c[6] = b[2]; c[7] = b[3];
    return __builtin_bit_cast(v16bf, c);
}

// A fragment from a row-major bf16 LDS tile (two contiguous 16B runs per lane).
__device__ __forceinline__ v16bf load_a_frag_rm(const unsigned short* p, int ld, int lane) {
    const unsigned short* r = p + (lane & 15) * ld + ((lane >> 4) << 3);
    v4u a = *(const v4u*)(r);        // k = 8*half + 0..7
    v4u b = *(const v4u*)(r + 16);   // k = 16 + 8*half + 0..7
    v8u c;
    c[0] = a[0]; c[1] = a[1]; c[2] = a[2]; c[3] = a[3];
    c[4] = b[0]; c[5] = b[1]; c[6] = b[2]; c[7] = b[3];
    return __builtin_bit_cast(v16bf, c);
}

__device__ __forceinline__ v8f wmma_bf16(v16bf a, v16bf b, v8f c) {
    return __builtin_amdgcn_wmma_f32_16x16x32_bf16(false, a, false, b, (short)0, c, false, false);
}

// ---------------- Kernel 0a: pack x (f32 -> bf16, A-operand fragment layout) --
// One thread per 8 contiguous k: 8 coalesced f32 reads -> one b128 store.
__global__ __launch_bounds__(256) void pack_x(const float* __restrict__ x,
                                              unsigned short* __restrict__ Xsw) {
    size_t t = (size_t)blockIdx.x * 256 + threadIdx.x;   // N*D/8 threads
    int m  = (int)(t >> 7);          // 128 groups of 8 per row
    int k8 = (int)(t & 127) << 3;
    const float* src = x + (size_t)m * D_DIM + k8;
    float4 u = *(const float4*)src;
    float4 v = *(const float4*)(src + 4);
    int mb = m >> 4, kb = k8 >> 5;
    int lane = (m & 15) + (((k8 >> 3) & 1) << 4);
    int run  = (k8 >> 4) & 1;
    size_t idx = ((size_t)mb * 32 + kb) * 512 + (size_t)(run * 32 + lane) * 8;
    *(uint4*)&Xsw[idx] =
        make_uint4(pack2(u.x, u.y), pack2(u.z, u.w), pack2(v.x, v.y), pack2(v.z, v.w));
}

// ---------------- Kernel 0b: pack W (f32 -> bf16, B-operand fragment layout) --
// One thread per 8-deep k-column (coalesced across n): -> one b128 store.
__global__ __launch_bounds__(256) void pack_w(
    const float* __restrict__ Wq, const float* __restrict__ Wk, const float* __restrict__ Wv,
    unsigned short* __restrict__ W0, unsigned short* __restrict__ W1,
    unsigned short* __restrict__ W2) {
    const int z = blockIdx.z;
    const float* W = (z == 0) ? Wq : ((z == 1) ? Wk : Wv);
    unsigned short* dst = (z == 0) ? W0 : ((z == 1) ? W1 : W2);
    size_t t = (size_t)blockIdx.x * 256 + threadIdx.x;   // D*D/8 threads
    int n  = (int)(t & 1023);
    int k8 = (int)(t >> 10) << 3;
    const float* src = W + (size_t)k8 * D_DIM + n;
    float f0 = src[0];
    float f1 = src[(size_t)1 * D_DIM];
    float f2 = src[(size_t)2 * D_DIM];
    float f3 = src[(size_t)3 * D_DIM];
    float f4 = src[(size_t)4 * D_DIM];
    float f5 = src[(size_t)5 * D_DIM];
    float f6 = src[(size_t)6 * D_DIM];
    float f7 = src[(size_t)7 * D_DIM];
    int kb = k8 >> 5, nb = n >> 4;
    int lane = (n & 15) + (((k8 >> 3) & 1) << 4);
    int run  = (k8 >> 4) & 1;
    size_t idx = ((size_t)kb * 64 + nb) * 512 + (size_t)(run * 32 + lane) * 8;
    *(uint4*)&dst[idx] =
        make_uint4(pack2(f0, f1), pack2(f2, f3), pack2(f4, f5), pack2(f6, f7));
}

// ---------------- Kernel 1: Q/K/V projections (all-bf16, pure WMMA stream) ----
// Block tile 128x128, 8 waves in 4x2, each wave 32x64 (2 A-frags x 4 B-frags,
// 8 WMMAs per k-step). No LDS, no barriers: operands stream from L2-resident
// fragment-native buffers as b128 loads.
// Outputs (fragment-native):
//  z=0: Qsw A-op, frag(mb,kb) at (mb*32+kb)*512          mb=row/16, kb=d/32
//  z=1: Ksw B-op (k=d, n=key), frag at (kb*256+nb)*512   kb=d/32,  nb=key/16
//  z=2: Vsw B-op (k=key, n=col), frag at (kb*64+nb)*512  kb=key/32, nb=col/16
__global__ __launch_bounds__(256) void qkv_gemm(
    const unsigned short* __restrict__ Xsw,
    const unsigned short* __restrict__ W0, const unsigned short* __restrict__ W1,
    const unsigned short* __restrict__ W2,
    unsigned short* __restrict__ Qsw, unsigned short* __restrict__ Ksw,
    unsigned short* __restrict__ Vsw) {
    const int tid  = threadIdx.x;
    const int lane = tid & 31;
    const int w    = tid >> 5;
    const int wm   = (w >> 1) * 32;
    const int wn   = (w & 1) * 64;
    const int z    = blockIdx.z;
    const unsigned short* Wsw = (z == 0) ? W0 : ((z == 1) ? W1 : W2);
    const int m0 = blockIdx.y * 128;
    const int n0 = blockIdx.x * 128;

    v8f acc[2][4];
#pragma unroll
    for (int i = 0; i < 2; ++i)
#pragma unroll
        for (int j = 0; j < 4; ++j)
#pragma unroll
            for (int r = 0; r < 8; ++r) acc[i][j][r] = 0.f;

    const size_t mb0 = (size_t)((m0 + wm) >> 4);
    const size_t nb0 = (size_t)((n0 + wn) >> 4);
#pragma unroll 4
    for (int kb = 0; kb < 32; ++kb) {
        v16bf a0 = load_frag_sw(Xsw + ((mb0 + 0) * 32 + kb) * 512, lane);
        v16bf a1 = load_frag_sw(Xsw + ((mb0 + 1) * 32 + kb) * 512, lane);
#pragma unroll
        for (int j = 0; j < 4; ++j) {
            v16bf b = load_frag_sw(Wsw + ((size_t)kb * 64 + nb0 + j) * 512, lane);
            acc[0][j] = wmma_bf16(a0, b, acc[0][j]);
            acc[1][j] = wmma_bf16(a1, b, acc[1][j]);
        }
    }

    // epilogue: scatter bf16 into fragment-native global layouts
    const int g = lane >> 4, cl = lane & 15;
#pragma unroll
    for (int i = 0; i < 2; ++i)
#pragma unroll
        for (int j = 0; j < 4; ++j)
#pragma unroll
            for (int r = 0; r < 8; ++r) {
                int R = m0 + wm + i * 16 + r + g * 8;   // token row
                int C = n0 + wn + j * 16 + cl;          // feature col
                unsigned short hv = f2bf(acc[i][j][r]);
                if (z == 0) {            // Q as A-op: m=R, k=C
                    size_t idx = ((size_t)(R >> 4) * 32 + (C >> 5)) * 512 +
                                 (size_t)((((C >> 4) & 1) * 32 + (R & 15) + (((C >> 3) & 1) << 4)) * 8) +
                                 (C & 7);
                    Qsw[idx] = hv;
                } else if (z == 1) {     // K as B-op: k=C (d), n=R (key)
                    size_t idx = ((size_t)(C >> 5) * 256 + (R >> 4)) * 512 +
                                 (size_t)((((C >> 4) & 1) * 32 + (R & 15) + (((C >> 3) & 1) << 4)) * 8) +
                                 (C & 7);
                    Ksw[idx] = hv;
                } else {                 // V as B-op: k=R (key), n=C (col)
                    size_t idx = ((size_t)(R >> 5) * 64 + (C >> 4)) * 512 +
                                 (size_t)((((R >> 4) & 1) * 32 + (C & 15) + (((R >> 3) & 1) << 4)) * 8) +
                                 (R & 7);
                    Vsw[idx] = hv;
                }
            }
}

// ---------------- Kernel 2: flash attention ----------------
// 64 Q rows / block; 8 waves each own a 64x128 column slice of O (32 v8f acc).
// S-phase streams Q/K fragments from L2 (2x global_load_b128 each): no LDS
// traffic and no barriers in the d-loop. Online softmax per 64-key block,
// then O += P*V with P (bf16) as LDS A-frags and V as global B-frags.
__global__ __launch_bounds__(256) void flash_attn(
    const unsigned short* __restrict__ Qsw, const unsigned short* __restrict__ Ksw,
    const unsigned short* __restrict__ Vsw, float* __restrict__ out) {
    __shared__ float          Sb[64][65];
    __shared__ unsigned short Pb[64][72];
    __shared__ float mrow[64], lrow[64], arow[64];

    const int tid  = threadIdx.x;
    const int lane = tid & 31;
    const int w    = tid >> 5;
    const int g    = lane >> 4, cl = lane & 15;
    const int q0   = blockIdx.x * 64;
    const float scale = 0.03125f;   // 1/sqrt(1024) = 1/sqrt(d_in)

    v8f oacc[4][8];
#pragma unroll
    for (int mi = 0; mi < 4; ++mi)
#pragma unroll
        for (int nj = 0; nj < 8; ++nj)
#pragma unroll
            for (int r = 0; r < 8; ++r) oacc[mi][nj][r] = 0.f;

    if (tid < 64) { mrow[tid] = -1e30f; lrow[tid] = 0.f; }
    __syncthreads();

    const int mi_s = w >> 1;        // this wave's S fragment row (0..3)
    const int ni0  = (w & 1) * 2;   // this wave's S fragment cols {ni0, ni0+1}
    const size_t mb = (size_t)(q0 >> 4) + mi_s;

    for (int j0 = 0; j0 < N_TOK; j0 += 64) {
        v8f sacc[2];
#pragma unroll
        for (int t = 0; t < 2; ++t)
#pragma unroll
            for (int r = 0; r < 8; ++r) sacc[t][r] = 0.f;

        const size_t nbk = (size_t)(j0 >> 4) + ni0;
#pragma unroll 4
        for (int kb = 0; kb < 32; ++kb) {
            v16bf aq = load_frag_sw(Qsw + (mb * 32 + kb) * 512, lane);
            v16bf b0 = load_frag_sw(Ksw + ((size_t)kb * 256 + nbk) * 512, lane);
            v16bf b1 = load_frag_sw(Ksw + ((size_t)kb * 256 + nbk + 1) * 512, lane);
            sacc[0] = wmma_bf16(aq, b0, sacc[0]);
            sacc[1] = wmma_bf16(aq, b1, sacc[1]);
        }
        // spill S fragments (C-layout: row = 16*mi + r + 8*half, col = 16*ni + lane%16)
#pragma unroll
        for (int t = 0; t < 2; ++t)
#pragma unroll
            for (int r = 0; r < 8; ++r)
                Sb[mi_s * 16 + r + g * 8][(ni0 + t) * 16 + cl] = sacc[t][r];
        __syncthreads();

        // online softmax, one thread per row
        if (tid < 64) {
            int r = tid;
            float mo = mrow[r], mx = mo;
            for (int c = 0; c < 64; ++c) mx = fmaxf(mx, Sb[r][c] * scale);
            float alpha = __expf(mo - mx);
            float sum = 0.f;
            for (int c = 0; c < 64; ++c) {
                float p = __expf(Sb[r][c] * scale - mx);
                sum += p;
                Pb[r][c] = f2bf(p);
            }
            mrow[r] = mx;
            lrow[r] = lrow[r] * alpha + sum;
            arow[r] = alpha;
        }
        __syncthreads();

        // rescale running O by alpha(row)
#pragma unroll
        for (int mi = 0; mi < 4; ++mi)
#pragma unroll
            for (int r = 0; r < 8; ++r) {
                float al = arow[mi * 16 + r + g * 8];
#pragma unroll
                for (int nj = 0; nj < 8; ++nj) oacc[mi][nj][r] *= al;
            }

        // O += P * V  (P from LDS as b128 A-frags, V as b128 B-frags from L2)
#pragma unroll
        for (int kk = 0; kk < 64; kk += 32) {
            v16bf ap[4];
#pragma unroll
            for (int mi = 0; mi < 4; ++mi) ap[mi] = load_a_frag_rm(&Pb[mi * 16][kk], 72, lane);
            const size_t vkb = (size_t)((j0 + kk) >> 5) * 64;
#pragma unroll
            for (int nj = 0; nj < 8; ++nj) {
                v16bf bv = load_frag_sw(Vsw + (vkb + (w * 8 + nj)) * 512, lane);
#pragma unroll
                for (int mi = 0; mi < 4; ++mi) oacc[mi][nj] = wmma_bf16(ap[mi], bv, oacc[mi][nj]);
            }
        }
        __syncthreads();
    }

    // epilogue: divide by l and write f32 output
#pragma unroll
    for (int mi = 0; mi < 4; ++mi)
#pragma unroll
        for (int nj = 0; nj < 8; ++nj)
#pragma unroll
            for (int r = 0; r < 8; ++r) {
                int row = mi * 16 + r + g * 8;
                float inv = 1.0f / lrow[row];
                out[(size_t)(q0 + row) * D_DIM + w * 128 + nj * 16 + cl] =
                    oacc[mi][nj][r] * inv;
            }
}

extern "C" void kernel_launch(void* const* d_in, const int* in_sizes, int n_in,
                              void* d_out, int out_size, void* d_ws, size_t ws_size,
                              hipStream_t stream) {
    const float* x  = (const float*)d_in[0];
    const float* Wq = (const float*)d_in[1];
    const float* Wk = (const float*)d_in[2];
    const float* Wv = (const float*)d_in[3];

    unsigned short* Qsw = (unsigned short*)d_ws;                  // 8 MB
    unsigned short* Ksw = Qsw + (size_t)N_TOK * D_DIM;            // 8 MB
    unsigned short* Vsw = Ksw + (size_t)N_TOK * D_DIM;            // 8 MB
    unsigned short* Xsw = Vsw + (size_t)N_TOK * D_DIM;            // 8 MB
    unsigned short* W0  = Xsw + (size_t)N_TOK * D_DIM;            // 2 MB
    unsigned short* W1  = W0 + (size_t)D_DIM * D_DIM;
    unsigned short* W2  = W1 + (size_t)D_DIM * D_DIM;
    float* out = (float*)d_out;

    pack_x<<<(N_TOK * (D_DIM / 8)) / 256, 256, 0, stream>>>(x, Xsw);
    dim3 gw((D_DIM * (D_DIM / 8)) / 256, 1, 3);
    pack_w<<<gw, 256, 0, stream>>>(Wq, Wk, Wv, W0, W1, W2);
    dim3 gq(D_DIM / 128, N_TOK / 128, 3);
    qkv_gemm<<<gq, 256, 0, stream>>>(Xsw, W0, W1, W2, Qsw, Ksw, Vsw);
    flash_attn<<<N_TOK / 64, 256, 0, stream>>>(Qsw, Ksw, Vsw, out);
}
